// MyMultiHeadAttention_88210038325405
// MI455X (gfx1250) — matrix operand order repaired and checked by reading, not compiled, not run on hardware
//
#include <hip/hip_runtime.h>
#include <hip/hip_bf16.h>

#define EMB  1024
#define NHEADS 16
#define HD   64
#define TGT  2048
#define SRCL 2048
#define BS   4

typedef __attribute__((ext_vector_type(16))) _Float16 v16h;
typedef __attribute__((ext_vector_type(8)))  _Float16 v8h;
typedef __attribute__((ext_vector_type(8)))  float    v8f;

// ---- fragment loaders (NT form: both operands stored K-contiguous, row-major) ----
// A matrix 16x32 f16: half-wave K-interleave in chunks of 8 (K0-7,16-23 | K8-15,24-31)
static __device__ __forceinline__ v16h load_fragA(const _Float16* __restrict__ p0, int ld) {
  const int lane = threadIdx.x & 31;
  const _Float16* p = p0 + (size_t)(lane & 15) * ld + ((lane >> 4) << 3);
  union { v16h v; v8h h[2]; } u;
  u.h[0] = *(const v8h*)(p);
  u.h[1] = *(const v8h*)(p + 16);
  return u.v;
}
// B matrix 32x16 f16 (given as its transpose, N-major/K-contig): half-wave K-interleave in chunks of 8 (K0-7,16-23 | K8-15,24-31)
static __device__ __forceinline__ v16h load_fragB(const _Float16* __restrict__ p0, int ld) {
  const int lane = threadIdx.x & 31;
  const _Float16* p = p0 + (size_t)(lane & 15) * ld + ((lane >> 4) << 3);
  union { v16h v; v8h h[2]; } u;
  u.h[0] = *(const v8h*)(p);
  u.h[1] = *(const v8h*)(p + 16);
  return u.v;
}

static __device__ __forceinline__ v8f wmma_f16(v16h a, v16h b, v8f c) {
  return __builtin_amdgcn_wmma_f32_16x16x32_f16(false, a, false, b, (short)0, c, false, false);
}

#define VZERO {0.f,0.f,0.f,0.f,0.f,0.f,0.f,0.f}

// ---------------- conversion: 8 elems/thread, b128 in / b128 out ----------------
__global__ void mha_cvt_f32_f16_v8(const float* __restrict__ src, _Float16* __restrict__ dst,
                                   int n8, float scale) {
  int i = blockIdx.x * blockDim.x + threadIdx.x;
  if (i >= n8) return;
  const float* s = src + (size_t)i * 8;
  const float4 a = *(const float4*)(s);
  const float4 b = *(const float4*)(s + 4);
  v8h o;
  o[0] = (_Float16)(a.x * scale); o[1] = (_Float16)(a.y * scale);
  o[2] = (_Float16)(a.z * scale); o[3] = (_Float16)(a.w * scale);
  o[4] = (_Float16)(b.x * scale); o[5] = (_Float16)(b.y * scale);
  o[6] = (_Float16)(b.z * scale); o[7] = (_Float16)(b.w * scale);
  *(v8h*)(dst + (size_t)i * 8) = o;
}

// ---------------- V transpose via LDS tile: [s,(b,h,d)] -> [b,h,d,s] ----------------
__global__ void __launch_bounds__(256)
mha_transpose_v(const _Float16* __restrict__ vh, _Float16* __restrict__ Vt) {
  __shared__ __align__(16) _Float16 tile[64][72];   // padded rows (144B, 16B-aligned)
  const int nSB = SRCL / 64;                        // 32
  const int st = blockIdx.x % nSB;
  const int bh = blockIdx.x / nSB;                  // b*NHEADS + h
  const int h = bh % NHEADS;
  const int b = bh / NHEADS;
  const int s0 = st * 64;
  const int tid = threadIdx.x;
#pragma unroll
  for (int p = 0; p < 2; ++p) {                     // coalesced read: d contiguous
    const int c = tid + 256 * p;                    // 0..511
    const int sl = c >> 3;
    const int off = (c & 7) << 3;
    *(v8h*)&tile[sl][off] =
        *(const v8h*)(vh + ((size_t)(s0 + sl) * BS + b) * EMB + h * HD + off);
  }
  __syncthreads();
#pragma unroll
  for (int p = 0; p < 2; ++p) {                     // coalesced write: s contiguous
    const int c = tid + 256 * p;
    const int dl = c >> 3;
    const int off = (c & 7) << 3;
    v8h o;
#pragma unroll
    for (int q = 0; q < 8; ++q) o[q] = tile[off + q][dl];
    *(v8h*)(Vt + ((size_t)bh * HD + dl) * SRCL + s0 + off) = o;
  }
}

// ---------------- NT GEMM: C[m,n] = sum_k A[m,k] B[n,k] ----------------
__global__ void __launch_bounds__(256)
mha_gemm_nt_f16(const _Float16* __restrict__ A, const _Float16* __restrict__ B,
                _Float16* __restrict__ C, int M, int N, int K) {
  const int lane = threadIdx.x & 31;
  const int wave = threadIdx.x >> 5;
  const int tilesN = N >> 6;
  const int tile = blockIdx.x * 8 + wave;
  const int tm = (tile / tilesN) << 6;
  const int tn = (tile % tilesN) << 6;
  const v8f vz = VZERO;
  v8f acc[4][4];
#pragma unroll
  for (int i = 0; i < 4; ++i)
#pragma unroll
    for (int j = 0; j < 4; ++j) acc[i][j] = vz;
  for (int k = 0; k < K; k += 32) {
    v16h a[4], b[4];
#pragma unroll
    for (int i = 0; i < 4; ++i) a[i] = load_fragA(A + (size_t)(tm + 16 * i) * K + k, K);
#pragma unroll
    for (int j = 0; j < 4; ++j) b[j] = load_fragB(B + (size_t)(tn + 16 * j) * K + k, K);
#pragma unroll
    for (int i = 0; i < 4; ++i)
#pragma unroll
      for (int j = 0; j < 4; ++j) acc[i][j] = wmma_f16(a[i], b[j], acc[i][j]);
  }
  const int r0 = tm + ((lane >> 4) << 3);
  const int c0 = tn + (lane & 15);
#pragma unroll
  for (int i = 0; i < 4; ++i)
#pragma unroll
    for (int j = 0; j < 4; ++j)
#pragma unroll
      for (int v = 0; v < 8; ++v)
        C[(size_t)(r0 + 16 * i + v) * N + (c0 + 16 * j)] = (_Float16)acc[i][j][v];
}

__global__ void __launch_bounds__(256)
mha_gemm_nt_f32_bias(const _Float16* __restrict__ A, const _Float16* __restrict__ B,
                     float* __restrict__ C, const float* __restrict__ bias,
                     int M, int N, int K) {
  const int lane = threadIdx.x & 31;
  const int wave = threadIdx.x >> 5;
  const int tilesN = N >> 6;
  const int tile = blockIdx.x * 8 + wave;
  const int tm = (tile / tilesN) << 6;
  const int tn = (tile % tilesN) << 6;
  const v8f vz = VZERO;
  v8f acc[4][4];
#pragma unroll
  for (int i = 0; i < 4; ++i)
#pragma unroll
    for (int j = 0; j < 4; ++j) acc[i][j] = vz;
  for (int k = 0; k < K; k += 32) {
    v16h a[4], b[4];
#pragma unroll
    for (int i = 0; i < 4; ++i) a[i] = load_fragA(A + (size_t)(tm + 16 * i) * K + k, K);
#pragma unroll
    for (int j = 0; j < 4; ++j) b[j] = load_fragB(B + (size_t)(tn + 16 * j) * K + k, K);
#pragma unroll
    for (int i = 0; i < 4; ++i)
#pragma unroll
      for (int j = 0; j < 4; ++j) acc[i][j] = wmma_f16(a[i], b[j], acc[i][j]);
  }
  const int r0 = tm + ((lane >> 4) << 3);
  const int c0 = tn + (lane & 15);
#pragma unroll
  for (int i = 0; i < 4; ++i)
#pragma unroll
    for (int j = 0; j < 4; ++j) {
      const float bb = bias[c0 + 16 * j];
#pragma unroll
      for (int v = 0; v < 8; ++v)
        C[(size_t)(r0 + 16 * i + v) * N + (c0 + 16 * j)] = acc[i][j][v] + bb;
    }
}

// ---------------- flash attention fwd: LDS-staged K/V shared by 4 waves ----------------
__global__ void __launch_bounds__(128)
mha_flash_fwd(const _Float16* __restrict__ Qh, const _Float16* __restrict__ Kh,
              const _Float16* __restrict__ Vt, _Float16* __restrict__ Oh,
              float* __restrict__ Ml, float* __restrict__ Ll) {
  __shared__ __align__(16) _Float16 ldsK[32 * 64];     // [s_local, d]
  __shared__ __align__(16) _Float16 ldsV[64 * 32];     // [d, s_local]
  __shared__ __align__(16) _Float16 ldsP[4][32 * 32];  // per-wave P tile
  const int tid = threadIdx.x;
  const int lane = tid & 31;
  const int wave = tid >> 5;
  const int nTB = TGT / 128;
  const int tb = blockIdx.x % nTB;
  const int bh = blockIdx.x / nTB;
  const int h = bh % NHEADS;
  const int b = bh / NHEADS;
  const int t0 = tb * 128 + wave * 32;
  const int ldq = BS * EMB;

  const _Float16* Qb = Qh + (size_t)b * EMB + h * HD;
  const _Float16* Kb = Kh + (size_t)b * EMB + h * HD;
  const _Float16* Vb = Vt + (size_t)(b * NHEADS + h) * HD * SRCL;

  v16h qf[2][2];
#pragma unroll
  for (int i = 0; i < 2; ++i)
#pragma unroll
    for (int kc = 0; kc < 2; ++kc)
      qf[i][kc] = load_fragA(Qb + (size_t)(t0 + 16 * i) * ldq + kc * 32, ldq);

  const v8f vz = VZERO;
  v8f o[2][4];
  float m_r[2][8], l_r[2][8];
#pragma unroll
  for (int i = 0; i < 2; ++i) {
#pragma unroll
    for (int j = 0; j < 4; ++j) o[i][j] = vz;
#pragma unroll
    for (int v = 0; v < 8; ++v) { m_r[i][v] = -1.0e30f; l_r[i][v] = 0.f; }
  }

  for (int s0 = 0; s0 < SRCL; s0 += 32) {
    __syncthreads();  // protect ldsK/ldsV overwrite vs previous PV reads
    // stage K (32x64) and V^T (64x32) tiles, 16B chunks, coalesced; prefetch next tile
#pragma unroll
    for (int p = 0; p < 2; ++p) {
      const int c = tid + 128 * p;             // 0..255
      const int kr = c >> 3, ko = (c & 7) << 3;
      *(v8h*)&ldsK[kr * 64 + ko] = *(const v8h*)(Kb + (size_t)(s0 + kr) * ldq + ko);
      const int vr = c >> 2, vo = (c & 3) << 3;
      *(v8h*)&ldsV[vr * 32 + vo] = *(const v8h*)(Vb + (size_t)vr * SRCL + s0 + vo);
      if (s0 + 32 < SRCL) {
        __builtin_prefetch(Kb + (size_t)(s0 + 32 + kr) * ldq + ko, 0, 3);
        __builtin_prefetch(Vb + (size_t)vr * SRCL + s0 + 32 + vo, 0, 3);
      }
    }
    __syncthreads();

    v8f sc[2][2];
#pragma unroll
    for (int i = 0; i < 2; ++i) { sc[i][0] = vz; sc[i][1] = vz; }
#pragma unroll
    for (int kc = 0; kc < 2; ++kc) {
      v16h kf0 = load_fragB(&ldsK[kc * 32], 64);
      v16h kf1 = load_fragB(&ldsK[16 * 64 + kc * 32], 64);
#pragma unroll
      for (int i = 0; i < 2; ++i) {
        sc[i][0] = wmma_f16(qf[i][kc], kf0, sc[i][0]);
        sc[i][1] = wmma_f16(qf[i][kc], kf1, sc[i][1]);
      }
    }
    // online softmax: C-slot v holds row 16i+v+8*(lane>=16); cols live across 16-lane group
#pragma unroll
    for (int i = 0; i < 2; ++i) {
#pragma unroll
      for (int v = 0; v < 8; ++v) {
        float mx = fmaxf(sc[i][0][v], sc[i][1][v]);
        mx = fmaxf(mx, __shfl_xor(mx, 1));
        mx = fmaxf(mx, __shfl_xor(mx, 2));
        mx = fmaxf(mx, __shfl_xor(mx, 4));
        mx = fmaxf(mx, __shfl_xor(mx, 8));
        const float mnew = fmaxf(m_r[i][v], mx);
        const float alpha = __expf(m_r[i][v] - mnew);
        m_r[i][v] = mnew;
        const float p0 = __expf(sc[i][0][v] - mnew);
        const float p1 = __expf(sc[i][1][v] - mnew);
        float ps = p0 + p1;
        ps += __shfl_xor(ps, 1);
        ps += __shfl_xor(ps, 2);
        ps += __shfl_xor(ps, 4);
        ps += __shfl_xor(ps, 8);
        l_r[i][v] = l_r[i][v] * alpha + ps;
        const int row = 16 * i + v + ((lane >> 4) << 3);
        ldsP[wave][row * 32 + (lane & 15)] = (_Float16)p0;
        ldsP[wave][row * 32 + 16 + (lane & 15)] = (_Float16)p1;
#pragma unroll
        for (int j = 0; j < 4; ++j) o[i][j][v] *= alpha;
      }
    }
    __syncthreads();
    v16h pf[2];
#pragma unroll
    for (int i = 0; i < 2; ++i) pf[i] = load_fragA(&ldsP[wave][16 * i * 32], 32);
#pragma unroll
    for (int j = 0; j < 4; ++j) {
      v16h vf = load_fragB(&ldsV[16 * j * 32], 32);
#pragma unroll
      for (int i = 0; i < 2; ++i) o[i][j] = wmma_f16(pf[i], vf, o[i][j]);
    }
  }
  // epilogue: normalize, store O as out-proj A matrix; write final (m,l)
#pragma unroll
  for (int i = 0; i < 2; ++i)
#pragma unroll
    for (int v = 0; v < 8; ++v) {
      const int t = t0 + 16 * i + v + ((lane >> 4) << 3);
      const float inv = 1.0f / l_r[i][v];
#pragma unroll
      for (int j = 0; j < 4; ++j)
        Oh[(size_t)(t * BS + b) * EMB + h * HD + 16 * j + (lane & 15)] =
            (_Float16)(o[i][j][v] * inv);
      if ((lane & 15) == 0) {
        const size_t idx = (size_t)(b * NHEADS + h) * TGT + t;
        Ml[idx] = m_r[i][v];
        Ll[idx] = l_r[i][v];
      }
    }
}

// ---------------- avg_weights: recompute scores per head with exact (m,l) ----------------
__global__ void __launch_bounds__(128)
mha_avg_probs(const _Float16* __restrict__ Qh, const _Float16* __restrict__ Kh,
              const float* __restrict__ Ml, const float* __restrict__ Ll,
              float* __restrict__ avg) {
  const int lane = threadIdx.x & 31;
  const int wave = threadIdx.x >> 5;
  const int nSB = SRCL / 32;
  const int nTB = TGT / 128;
  const int sb = blockIdx.x % nSB;
  const int tb = (blockIdx.x / nSB) % nTB;
  const int b = blockIdx.x / (nSB * nTB);
  const int t0 = tb * 128 + wave * 32;
  const int s0 = sb * 32;
  const int ldq = BS * EMB;
  const v8f vz = VZERO;
  v8f acc[2][2];
#pragma unroll
  for (int i = 0; i < 2; ++i) { acc[i][0] = vz; acc[i][1] = vz; }
  for (int h = 0; h < NHEADS; ++h) {
    const _Float16* Qb = Qh + (size_t)b * EMB + h * HD;
    const _Float16* Kb = Kh + (size_t)b * EMB + h * HD;
    v8f sc[2][2];
#pragma unroll
    for (int i = 0; i < 2; ++i) { sc[i][0] = vz; sc[i][1] = vz; }
#pragma unroll
    for (int kc = 0; kc < 2; ++kc) {
      v16h kf0 = load_fragB(Kb + (size_t)(s0)*ldq + kc * 32, ldq);
      v16h kf1 = load_fragB(Kb + (size_t)(s0 + 16) * ldq + kc * 32, ldq);
#pragma unroll
      for (int i = 0; i < 2; ++i) {
        v16h qf = load_fragA(Qb + (size_t)(t0 + 16 * i) * ldq + kc * 32, ldq);
        sc[i][0] = wmma_f16(qf, kf0, sc[i][0]);
        sc[i][1] = wmma_f16(qf, kf1, sc[i][1]);
      }
    }
#pragma unroll
    for (int i = 0; i < 2; ++i)
#pragma unroll
      for (int v = 0; v < 8; ++v) {
        const int t = t0 + 16 * i + v + ((lane >> 4) << 3);
        const size_t idx = (size_t)(b * NHEADS + h) * TGT + t;
        const float m = Ml[idx];
        const float s = 0.0625f / Ll[idx];  // 1/H folded in
        acc[i][0][v] += __expf(sc[i][0][v] - m) * s;
        acc[i][1][v] += __expf(sc[i][1][v] - m) * s;
      }
  }
#pragma unroll
  for (int i = 0; i < 2; ++i)
#pragma unroll
    for (int v = 0; v < 8; ++v) {
      const int t = t0 + 16 * i + v + ((lane >> 4) << 3);
#pragma unroll
      for (int jj = 0; jj < 2; ++jj)
        avg[((size_t)b * TGT + t) * SRCL + s0 + 16 * jj + (lane & 15)] = acc[i][jj][v];
    }
}

// ---------------- host ----------------
extern "C" void kernel_launch(void* const* d_in, const int* in_sizes, int n_in,
                              void* d_out, int out_size, void* d_ws, size_t ws_size,
                              hipStream_t stream) {
  (void)in_sizes; (void)n_in; (void)out_size; (void)ws_size;
  const float* query = (const float*)d_in[0];
  const float* key   = (const float*)d_in[1];
  const float* value = (const float*)d_in[2];
  const float* w_q   = (const float*)d_in[3];
  const float* w_k   = (const float*)d_in[4];
  const float* w_v   = (const float*)d_in[5];
  const float* w_out = (const float*)d_in[6];
  const float* b_out = (const float*)d_in[7];

  float* Z   = (float*)d_out;                       // (T,B,E)
  float* avg = Z + (size_t)TGT * BS * EMB;          // (B,T,S)

  const size_t NX = (size_t)TGT * BS * EMB;         // 8388608
  const size_t NW = (size_t)EMB * EMB;              // 1048576
  _Float16* Xq = (_Float16*)d_ws;
  _Float16* Xk = Xq + NX;
  _Float16* Xv = Xk + NX;
  _Float16* qh = Xv + NX;
  _Float16* kh = qh + NX;
  _Float16* vh = kh + NX;
  _Float16* Vt = vh + NX;
  _Float16* Oh = Vt + NX;
  _Float16* Wq = Oh + NX;
  _Float16* Wk = Wq + NW;
  _Float16* Wv = Wk + NW;
  _Float16* Wo = Wv + NW;
  float* Ml = (float*)(Wo + NW);
  float* Ll = Ml + (size_t)BS * NHEADS * TGT;

  const int n8Big = (int)(NX / 8), n8W = (int)(NW / 8);
  const int gBig = (n8Big + 255) / 256, gW = (n8W + 255) / 256;

  // 1. convert inputs / weights to f16 (fold 1/sqrt(d)=0.125 into Wq)
  mha_cvt_f32_f16_v8<<<gBig, 256, 0, stream>>>(query, Xq, n8Big, 1.0f);
  mha_cvt_f32_f16_v8<<<gBig, 256, 0, stream>>>(key,   Xk, n8Big, 1.0f);
  mha_cvt_f32_f16_v8<<<gBig, 256, 0, stream>>>(value, Xv, n8Big, 1.0f);
  mha_cvt_f32_f16_v8<<<gW, 256, 0, stream>>>(w_q, Wq, n8W, 0.125f);
  mha_cvt_f32_f16_v8<<<gW, 256, 0, stream>>>(w_k, Wk, n8W, 1.0f);
  mha_cvt_f32_f16_v8<<<gW, 256, 0, stream>>>(w_v, Wv, n8W, 1.0f);
  mha_cvt_f32_f16_v8<<<gW, 256, 0, stream>>>(w_out, Wo, n8W, 1.0f);

  // 2. Q/K/V projections: [8192x1024] @ [1024x1024]^T, WMMA NT-GEMM
  const int M = TGT * BS, N = EMB, K = EMB;
  const int gemmBlocks = (M / 64) * (N / 64) / 8;   // 256
  mha_gemm_nt_f16<<<gemmBlocks, 256, 0, stream>>>(Xq, Wq, qh, M, N, K);
  mha_gemm_nt_f16<<<gemmBlocks, 256, 0, stream>>>(Xk, Wk, kh, M, N, K);
  mha_gemm_nt_f16<<<gemmBlocks, 256, 0, stream>>>(Xv, Wv, vh, M, N, K);

  // 3. transpose V to [b,h,d,s] so PV is NT-form (LDS-tiled, both sides coalesced)
  mha_transpose_v<<<BS * NHEADS * (SRCL / 64), 256, 0, stream>>>(vh, Vt);

  // 4. flash attention -> O (f16) + final (m,l) per row
  mha_flash_fwd<<<BS * NHEADS * (TGT / 128), 128, 0, stream>>>(qh, kh, Vt, Oh, Ml, Ll);

  // 5. averaged attention weights (recompute with exact m,l; no atomics)
  mha_avg_probs<<<BS * (TGT / 128) * (SRCL / 32), 128, 0, stream>>>(qh, kh, Ml, Ll, avg);

  // 6. output projection + bias -> Z (f32)
  mha_gemm_nt_f32_bias<<<gemmBlocks, 256, 0, stream>>>(Oh, Wo, Z, b_out, M, N, K);
}
